// Attention_12713103197044
// MI455X (gfx1250) — matrix operand order repaired and checked by reading, not compiled, not run on hardware
//
#include <hip/hip_runtime.h>

#define B_ 8
#define T_ 128
#define S_ 512
#define D_ 512
#define C_ 512

typedef __attribute__((ext_vector_type(16))) __bf16 v16bf;
typedef __attribute__((ext_vector_type(8)))  float  v8f;
typedef __attribute__((ext_vector_type(4)))  unsigned int u32x4;
typedef __attribute__((ext_vector_type(8)))  int          i32x8;
typedef __attribute__((ext_vector_type(4)))  int          i32x4;

#if __has_builtin(__builtin_amdgcn_tensor_load_to_lds) && __has_builtin(__builtin_amdgcn_s_wait_tensorcnt)
#define USE_TDM 1
#else
#define USE_TDM 0
#endif

__device__ __forceinline__ unsigned short f2bf(float f) {
  unsigned int u = __float_as_uint(f);
  u += 0x7FFFu + ((u >> 16) & 1u);           // RNE
  return (unsigned short)(u >> 16);
}

// tanh on the trans pipe: tanh(x) = 1 - 2/(exp2(2x*log2e)+1)
__device__ __forceinline__ float fast_tanh(float x) {
  float t = __builtin_amdgcn_exp2f(x * 2.8853900817779268f);
  return 1.0f - 2.0f * __builtin_amdgcn_rcpf(t + 1.0f);
}

union Frag { uint4 q[2]; v16bf v; };

#if USE_TDM
// TDM 2D tile load: 32(k, contiguous) x 64(rows) bf16 tile -> LDS, with TDM-side
// padding of 4 dwords per 16-dword row => LDS row stride 40 ushorts (80B).
// D# layout per CDNA5 ISA §8 (group0 128b, group1 256b; groups 2/3 zero => 2D).
// Toolchain here is the 6-arg clang-23 form:
//   (u32x4 g0, i32x8 g1, i32x4 g2, i32x4 g3, i32x8 zero, i32 cpol)
__device__ __forceinline__ void tdm_load_tile_2d(const unsigned short* gsrc,
                                                 unsigned int lds_byte_off,
                                                 unsigned int ld_elems)
{
  unsigned long long ga = (unsigned long long)gsrc;
  u32x4 g0;
  g0.x = 1u;                                        // count=1, user mode, no gather
  g0.y = lds_byte_off;                              // lds_addr
  g0.z = (unsigned int)ga;                          // global_addr[31:0]
  g0.w = (unsigned int)(ga >> 32) | 0x80000000u;    // global_addr[56:32] | type=2
  unsigned long long q0 = (1ull << 16)              // data_size = 1 -> 2 bytes
                        | (1ull << 20)              // pad_enable
                        | (3ull << 22)              // pad_interval: 16 dwords (one 64B row)
                        | (3ull << 25)              // pad_amount: 4 dwords (8 bf16)
                        | (32ull << 48);            // tensor_dim0[15:0] = 32
  unsigned long long q1 = 0ull                      // tensor_dim0[31:16]
                        | (64ull << 16)             // tensor_dim1 = 64
                        | (32ull << 48);            // tile_dim0 = 32
  unsigned long long q2 = 64ull                     // tile_dim1 = 64, tile_dim2 = 0
                        | ((unsigned long long)ld_elems << 32); // tensor_dim0_stride lo32
  unsigned long long q3 = 0ull;                     // stride hi=0, tensor_dim1_stride=0
  i32x8 g1;
  g1[0] = (int)(unsigned int)q0;  g1[1] = (int)(unsigned int)(q0 >> 32);
  g1[2] = (int)(unsigned int)q1;  g1[3] = (int)(unsigned int)(q1 >> 32);
  g1[4] = (int)(unsigned int)q2;  g1[5] = (int)(unsigned int)(q2 >> 32);
  g1[6] = (int)(unsigned int)q3;  g1[7] = (int)(unsigned int)(q3 >> 32);
  i32x4 z4 = {0, 0, 0, 0};
  i32x8 z8 = {0, 0, 0, 0, 0, 0, 0, 0};
  __builtin_amdgcn_tensor_load_to_lds(g0, g1, z4, z4, z8, 0);
}
__device__ __forceinline__ unsigned int lds_off(const void* p) {
  return (unsigned int)(unsigned long long)p;       // generic LDS addr: offset in low 32b
}
#endif

// C = [A0 | A1] @ Wt^T + bias; A row-major bf16 [M,Kx]; Wt row-major bf16 [N, K0+K1].
// Writes fp32 Cf and/or bf16 Cbf (both [M,N], element stride strC per batch z).
// Requires M%64==0 (grid), N%64==0, K0%32==0, K1%32==0.
__global__ __launch_bounds__(256)
void wmma_gemm_bf16_kernel(const unsigned short* __restrict__ A0,
                           const unsigned short* __restrict__ A1,
                           const unsigned short* __restrict__ Wt,
                           const float* __restrict__ bias,
                           float* __restrict__ Cf, unsigned short* __restrict__ Cbf,
                           int M, int N, int K0, int K1, int ldW,
                           long strA0, long strA1, long strW, long strC, int do_tanh)
{
  __shared__ __align__(16) unsigned short shA[2][64][40];
  __shared__ __align__(16) unsigned short shB[2][64][40];

  const int tid  = threadIdx.x;
  const int lane = tid & 31;
  const int wave = tid >> 5;
  const int wm   = wave >> 1;
  const int wn   = wave & 1;
  const int bm   = blockIdx.y * 64;
  const int bn   = blockIdx.x * 64;
  const int bz   = blockIdx.z;

  const unsigned short* A0b = A0 + (long)bz * strA0;
  const unsigned short* A1b = A1 ? (A1 + (long)bz * strA1) : nullptr;
  const unsigned short* WtB = Wt + (long)bz * strW;

  const int K = K0 + K1;
  v8f acc0 = {};
  v8f acc1 = {};

  const int row16 = wm * 16 + (lane & 15);
  const int klo   = (lane < 16) ? 0 : 8;
  const int kb    = (lane < 16) ? 0 : 16;
  const int n0    = wn * 32 + (lane & 15);

#if USE_TDM
  int cur = 0;
  if (wave == 0) {   // prime the pipeline: tile k=0 into buffer 0 (TDM, wave-scalar)
    tdm_load_tile_2d(A0b + (long)bm * K0, lds_off(&shA[0][0][0]), (unsigned)K0);
    tdm_load_tile_2d(WtB + (long)bn * ldW, lds_off(&shB[0][0][0]), (unsigned)ldW);
  }
  for (int k = 0; k < K; k += 32) {
    const bool more = (k + 32 < K);
    __syncthreads();                       // all readers done with buf[cur^1]
    if (wave == 0) {
      if (more) {
        const int kn = k + 32;
        const unsigned short* Ap; int ldA; int kc;
        if (kn < K0) { Ap = A0b; ldA = K0; kc = kn; }
        else         { Ap = A1b; ldA = K1; kc = kn - K0; }
        tdm_load_tile_2d(Ap + (long)bm * ldA + kc, lds_off(&shA[cur ^ 1][0][0]), (unsigned)ldA);
        tdm_load_tile_2d(WtB + (long)bn * ldW + kn, lds_off(&shB[cur ^ 1][0][0]), (unsigned)ldW);
        __builtin_amdgcn_s_wait_tensorcnt(2);   // current pair done (in-order), next in flight
      } else {
        __builtin_amdgcn_s_wait_tensorcnt(0);
      }
    }
    __syncthreads();                       // buf[cur] visible to all waves

    Frag fa, fb0, fb1;
    fa.q[0]  = *(const uint4*)&shA[cur][row16][klo];
    fa.q[1]  = *(const uint4*)&shA[cur][row16][16 + klo];
    fb0.q[0] = *(const uint4*)&shB[cur][n0][klo];
    fb0.q[1] = *(const uint4*)&shB[cur][n0][16 + klo];
    fb1.q[0] = *(const uint4*)&shB[cur][n0 + 16][klo];
    fb1.q[1] = *(const uint4*)&shB[cur][n0 + 16][16 + klo];
    acc0 = __builtin_amdgcn_wmma_f32_16x16x32_bf16(false, fa.v, false, fb0.v,
                                                   (short)0, acc0, false, false);
    acc1 = __builtin_amdgcn_wmma_f32_16x16x32_bf16(false, fa.v, false, fb1.v,
                                                   (short)0, acc1, false, false);
    cur ^= 1;
  }
#else
  const int arow = tid >> 2;
  const int aseg = (tid & 3) * 8;
  for (int k = 0; k < K; k += 32) {
    const unsigned short* Ap; int ldA; int kc;
    if (k < K0) { Ap = A0b; ldA = K0; kc = k; }
    else        { Ap = A1b; ldA = K1; kc = k - K0; }
    *(uint4*)&shA[0][arow][aseg] = *(const uint4*)(Ap + (long)(bm + arow) * ldA + kc + aseg);
    *(uint4*)&shB[0][arow][aseg] = *(const uint4*)(WtB + (long)(bn + arow) * ldW + k + aseg);
    if (k + 32 < K)
      __builtin_prefetch(WtB + (long)(bn + arow) * ldW + k + 32 + aseg, 0, 1);
    __syncthreads();

    Frag fa, fb0, fb1;
    fa.q[0]  = *(const uint4*)&shA[0][row16][klo];
    fa.q[1]  = *(const uint4*)&shA[0][row16][16 + klo];
    fb0.q[0] = *(const uint4*)&shB[0][n0][klo];
    fb0.q[1] = *(const uint4*)&shB[0][n0][16 + klo];
    fb1.q[0] = *(const uint4*)&shB[0][n0 + 16][klo];
    fb1.q[1] = *(const uint4*)&shB[0][n0 + 16][16 + klo];
    acc0 = __builtin_amdgcn_wmma_f32_16x16x32_bf16(false, fa.v, false, fb0.v,
                                                   (short)0, acc0, false, false);
    acc1 = __builtin_amdgcn_wmma_f32_16x16x32_bf16(false, fa.v, false, fb1.v,
                                                   (short)0, acc1, false, false);
    __syncthreads();
  }
#endif

  // Epilogue: C/D layout (VGPR r -> M = r + (lane>=16 ? 8 : 0), N = lane&15)
  const int rbase = bm + wm * 16 + ((lane >> 4) ? 8 : 0);
  const int c0 = bn + wn * 32 + (lane & 15);
  const int c1 = c0 + 16;
  const float b0v = bias ? bias[c0] : 0.0f;
  const float b1v = bias ? bias[c1] : 0.0f;
  float* Cfb = Cf ? (Cf + (long)bz * strC) : nullptr;
  unsigned short* Cbb = Cbf ? (Cbf + (long)bz * strC) : nullptr;
  #pragma unroll
  for (int r = 0; r < 8; ++r) {
    float v0 = acc0[r] + b0v;
    float v1 = acc1[r] + b1v;
    if (do_tanh) { v0 = fast_tanh(v0); v1 = fast_tanh(v1); }
    const long o0 = (long)(rbase + r) * N + c0;
    const long o1 = (long)(rbase + r) * N + c1;
    if (Cfb) { Cfb[o0] = v0; Cfb[o1] = v1; }
    if (Cbb) { Cbb[o0] = f2bf(v0); Cbb[o1] = f2bf(v1); }
  }
}

// fp32 -> bf16, 8 elements/thread; n % 2048 == 0
__global__ __launch_bounds__(256)
void cvt_bf16_kernel(const float* __restrict__ src, unsigned short* __restrict__ dst, long n)
{
  long i = ((long)blockIdx.x * 256 + threadIdx.x) * 8;
  if (i >= n) return;
  float4 v0 = *(const float4*)(src + i);
  float4 v1 = *(const float4*)(src + i + 4);
  union { unsigned short s[8]; uint4 q; } c;
  c.s[0] = f2bf(v0.x); c.s[1] = f2bf(v0.y); c.s[2] = f2bf(v0.z); c.s[3] = f2bf(v0.w);
  c.s[4] = f2bf(v1.x); c.s[5] = f2bf(v1.y); c.s[6] = f2bf(v1.z); c.s[7] = f2bf(v1.w);
  *(uint4*)(dst + i) = c.q;
}

// dst[z][c][r] = bf16(src[z][r][c]); R%32==0, Cc%32==0
__global__ __launch_bounds__(256)
void cvt_t_bf16_kernel(const float* __restrict__ src, unsigned short* __restrict__ dst,
                       int R, int Cc, long sS, long sD)
{
  __shared__ float t[32][33];
  const int r0 = blockIdx.y * 32, c0 = blockIdx.x * 32;
  const float* s = src + (long)blockIdx.z * sS;
  unsigned short* d = dst + (long)blockIdx.z * sD;
  const int tr = threadIdx.x >> 5, tc = threadIdx.x & 31;
  #pragma unroll
  for (int p = 0; p < 32; p += 8)
    t[tr + p][tc] = s[(long)(r0 + tr + p) * Cc + (c0 + tc)];
  __syncthreads();
  #pragma unroll
  for (int p = 0; p < 32; p += 8)
    d[(long)(c0 + tr + p) * R + (r0 + tc)] = f2bf(t[tc][tr + p]);
}

// One block per (b,t): logit[s] = sum_d tanh(mo+ma)*q + qb, softmax over S.
__global__ __launch_bounds__(256)
void logits_softmax_kernel(const float* __restrict__ mo, const float* __restrict__ ma,
                           const float* __restrict__ qw, const float* __restrict__ qb,
                           float* __restrict__ attn, unsigned short* __restrict__ attn_bf)
{
  __shared__ float smo[D_];
  __shared__ float sq[D_];
  __shared__ float slog[S_];
  __shared__ float red[256];

  const int tid  = threadIdx.x;
  const int bt   = blockIdx.x;
  const int b    = bt / T_;
  const int lane = tid & 31;
  const int wave = tid >> 5;

  for (int i = tid; i < D_; i += 256) {
    smo[i] = mo[(long)bt * D_ + i];
    sq[i]  = qw[i];
  }
  __syncthreads();

  const float* maB = ma + (long)b * S_ * D_;
  const float qb0 = qb[0];

  for (int s = wave; s < S_; s += 8) {
    const float4* r4 = (const float4*)(maB + (long)s * D_) + lane * 4;
    float acc = 0.f;
    #pragma unroll
    for (int i = 0; i < 4; ++i) {
      float4 v = r4[i];
      int d = lane * 16 + i * 4;
      acc += fast_tanh(smo[d + 0] + v.x) * sq[d + 0];
      acc += fast_tanh(smo[d + 1] + v.y) * sq[d + 1];
      acc += fast_tanh(smo[d + 2] + v.z) * sq[d + 2];
      acc += fast_tanh(smo[d + 3] + v.w) * sq[d + 3];
    }
    #pragma unroll
    for (int off = 16; off > 0; off >>= 1)
      acc += __shfl_xor(acc, off, 32);
    if (lane == 0) slog[s] = acc + qb0;
  }
  __syncthreads();

  float mx = -3.402823466e38f;
  for (int i = tid; i < S_; i += 256) mx = fmaxf(mx, slog[i]);
  red[tid] = mx; __syncthreads();
  for (int w = 128; w > 0; w >>= 1) {
    if (tid < w) red[tid] = fmaxf(red[tid], red[tid + w]);
    __syncthreads();
  }
  mx = red[0]; __syncthreads();

  float sum = 0.f;
  for (int i = tid; i < S_; i += 256) {
    float e = __builtin_amdgcn_exp2f((slog[i] - mx) * 1.4426950408889634f);
    slog[i] = e;
    sum += e;
  }
  red[tid] = sum; __syncthreads();
  for (int w = 128; w > 0; w >>= 1) {
    if (tid < w) red[tid] += red[tid + w];
    __syncthreads();
  }
  const float inv = 1.0f / red[0];

  for (int i = tid; i < S_; i += 256) {
    float v = slog[i] * inv;
    attn[(long)bt * S_ + i] = v;
    attn_bf[(long)bt * S_ + i] = f2bf(v);
  }
}

extern "C" void kernel_launch(void* const* d_in, const int* in_sizes, int n_in,
                              void* d_out, int out_size, void* d_ws, size_t ws_size,
                              hipStream_t stream)
{
  const float* output    = (const float*)d_in[0];  // [B,T,D]
  const float* context   = (const float*)d_in[1];  // [B,S,C]
  const float* dec_w_w   = (const float*)d_in[2];  // [D,D]
  const float* dec_w_b   = (const float*)d_in[3];
  const float* attn_w_w  = (const float*)d_in[4];  // [C,D]
  const float* attn_w_b  = (const float*)d_in[5];
  const float* query_w_w = (const float*)d_in[6];  // [D,1]
  const float* query_w_b = (const float*)d_in[7];  // [1]
  const float* out_w     = (const float*)d_in[8];  // [D+C, D]
  const float* out_b     = (const float*)d_in[9];

  float* out_final = (float*)d_out;                     // [B,T,D]
  float* attn_out  = out_final + (size_t)B_ * T_ * D_;  // [B,T,S]

  // workspace carve (fp32 then bf16 regions; all 16B aligned)
  float* mo = (float*)d_ws;                              // B*T*D
  float* ma = mo + (size_t)B_ * T_ * D_;                 // B*S*D
  unsigned short* u = (unsigned short*)(ma + (size_t)B_ * S_ * D_);
  unsigned short* output_bf  = u;  u += (size_t)B_ * T_ * D_;
  unsigned short* context_bf = u;  u += (size_t)B_ * S_ * C_;
  unsigned short* contextT   = u;  u += (size_t)B_ * C_ * S_;
  unsigned short* dec_wT     = u;  u += (size_t)D_ * D_;
  unsigned short* attn_wT    = u;  u += (size_t)D_ * C_;
  unsigned short* out_wT     = u;  u += (size_t)D_ * (D_ + C_);
  unsigned short* attn_bf    = u;  u += (size_t)B_ * T_ * S_;
  unsigned short* mix_bf     = u;  u += (size_t)B_ * T_ * C_;

  dim3 blk(256);

  // ---- one-time bf16 conversions / weight transposes ----
  cvt_bf16_kernel<<<dim3((B_*T_*D_) / 2048), blk, 0, stream>>>(output, output_bf, (long)B_*T_*D_);
  cvt_bf16_kernel<<<dim3((B_*S_*C_) / 2048), blk, 0, stream>>>(context, context_bf, (long)B_*S_*C_);
  cvt_t_bf16_kernel<<<dim3(D_/32, D_/32, 1), blk, 0, stream>>>(dec_w_w, dec_wT, D_, D_, 0, 0);
  cvt_t_bf16_kernel<<<dim3(D_/32, C_/32, 1), blk, 0, stream>>>(attn_w_w, attn_wT, C_, D_, 0, 0);
  cvt_t_bf16_kernel<<<dim3(D_/32, (D_+C_)/32, 1), blk, 0, stream>>>(out_w, out_wT, D_+C_, D_, 0, 0);
  cvt_t_bf16_kernel<<<dim3(C_/32, S_/32, B_), blk, 0, stream>>>(context, contextT, S_, C_,
                                                                (long)S_*C_, (long)C_*S_);

  // 1) mo = output @ dec_w + dec_b
  wmma_gemm_bf16_kernel<<<dim3(D_/64, (B_*T_)/64, 1), blk, 0, stream>>>(
      output_bf, nullptr, dec_wT, dec_w_b, mo, nullptr,
      B_*T_, D_, D_, 0, D_, 0, 0, 0, 0, 0);

  // 2) ma = context @ attn_w + attn_b
  wmma_gemm_bf16_kernel<<<dim3(D_/64, (B_*S_)/64, 1), blk, 0, stream>>>(
      context_bf, nullptr, attn_wT, attn_w_b, ma, nullptr,
      B_*S_, D_, C_, 0, C_, 0, 0, 0, 0, 0);

  // 3) fused tanh-dot logits + softmax -> attn (fp32 to d_out, bf16 for mix GEMM)
  logits_softmax_kernel<<<dim3(B_*T_), blk, 0, stream>>>(
      mo, ma, query_w_w, query_w_b, attn_out, attn_bf);

  // 4) mix[b] = attn[b] @ context[b]   (bf16 out only)
  wmma_gemm_bf16_kernel<<<dim3(C_/64, T_/64, B_), blk, 0, stream>>>(
      attn_bf, nullptr, contextT, nullptr, nullptr, mix_bf,
      T_, C_, S_, 0, S_, (long)T_*S_, 0, (long)C_*S_, (long)T_*C_, 0);

  // 5) out = tanh([mix | output] @ out_w + out_b)
  wmma_gemm_bf16_kernel<<<dim3(D_/64, (B_*T_)/64, 1), blk, 0, stream>>>(
      mix_bf, output_bf, out_wT, out_b, out_final, nullptr,
      B_*T_, D_, C_, D_, D_+C_, 0, 0, 0, 0, 1);
}
